// Semantic_Mapping_32280974197199
// MI455X (gfx1250) — compile-verified, matrix operand order checked by reading, not run on hardware
//
#include <hip/hip_runtime.h>
#include <hip/hip_bf16.h>

typedef _Float16 v16h __attribute__((ext_vector_type(16)));
typedef float    v8f  __attribute__((ext_vector_type(8)));

// ---------------- problem constants ----------------
#define BSZ    4
#define IMG_H  480
#define IMG_W  640
#define NCAT   16
#define NCH    20            // 4 + NCAT
#define MDIM   480           // map size M
#define VRD    100           // vision range (cells)
#define ZHD    80            // z bins
#define MINZI  9             // MIN_Z
#define MAXZI  49            // MAX_Z
#define AGENT_HF 155.0f
#define XCAMF  319.5f
#define ZCAMF  239.5f
#define DEGF   57.29577951308232f
#define PIF    3.14159265358979323846f
#define CY1    240           // crop rows [240,340)
#define CY2    340
#define CX1    190           // crop cols [190,290)
#define CX2    290

// ---------------- workspace layout (float units) ----------------
#define PARAMS_OFF 0                                   // 4 x {ct, st, stx, sty}
#define PREDS_OFF  16                                  // (b,18,100,100): 0=fp_map,1=fp_exp,2..17=cat
#define PREDS_SIZE (BSZ * 18 * VRD * VRD)
#define VOX_OFF    (PREDS_OFF + PREDS_SIZE)            // (b,17,y,x,z) f32
#define VOX_SIZE   (BSZ * 17 * VRD * VRD * ZHD)
#define ROT_OFF    (VOX_OFF + VOX_SIZE)                // rotated (b,20,480,480)
#define ROT_SIZE   (BSZ * NCH * MDIM * MDIM)

// ---------------- output layout (float units, concatenated tuple) ----------------
#define O_FP    0
#define O_MAP   (O_FP + BSZ * VRD * VRD)               // 40000
#define O_POSE1 (O_MAP + BSZ * NCH * MDIM * MDIM)      // 18,472,000
#define O_POSE2 (O_POSE1 + BSZ * 3)
#define O_TRANS (O_POSE2 + BSZ * 3)

// =====================================================================
// Kernel 1: poses + per-batch transform params
// =====================================================================
__global__ void sm_pose_kernel(const float* __restrict__ pose_obs,
                               const float* __restrict__ poses_last,
                               float* __restrict__ out,
                               float* __restrict__ ws_params) {
    int b = threadIdx.x;
    if (b >= BSZ) return;
    float px = poses_last[b * 3 + 0];
    float py = poses_last[b * 3 + 1];
    float pt = poses_last[b * 3 + 2];
    float o  = pt / DEGF;
    float so = __sinf(o), co = __cosf(o);
    float dx = pose_obs[b * 3 + 0];
    float dy = pose_obs[b * 3 + 1];
    float dt = pose_obs[b * 3 + 2];
    float yy = py + dx * so + dy * co;
    float xx = px + dx * co - dy * so;
    float tt = pt + dt * DEGF;
    tt = fmodf(tt - 180.0f, 360.0f) + 180.0f;
    tt = fmodf(tt + 180.0f, 360.0f) - 180.0f;

    out[O_POSE1 + b * 3 + 0] = xx;
    out[O_POSE1 + b * 3 + 1] = yy;
    out[O_POSE1 + b * 3 + 2] = tt;
    out[O_POSE2 + b * 3 + 0] = xx;
    out[O_POSE2 + b * 3 + 1] = yy;
    out[O_POSE2 + b * 3 + 2] = tt;

    const float half = (float)(MDIM / 2);
    float stx = -(xx * 100.0f / 5.0f - half) / half;
    float sty = -(yy * 100.0f / 5.0f - half) / half;
    float t   = (90.0f - tt) * PIF / 180.0f;
    ws_params[b * 4 + 0] = __cosf(t);
    ws_params[b * 4 + 1] = __sinf(t);
    ws_params[b * 4 + 2] = stx;
    ws_params[b * 4 + 3] = sty;
}

// =====================================================================
// Kernel 2: trilinear splat of 17 feature channels into voxel grid
// voxels layout: (b, c, y, x, z) f32 — already "swapaxes(2,3)" relative to ref
// =====================================================================
__global__ void sm_splat_kernel(const float* __restrict__ obs,
                                const float* __restrict__ view_angles,
                                float* __restrict__ vox) {
    long long idx = (long long)blockIdx.x * blockDim.x + threadIdx.x;
    const long long total = (long long)BSZ * IMG_H * IMG_W;
    if (idx >= total) return;
    int w = (int)(idx % IMG_W);
    int h = (int)((idx / IMG_W) % IMG_H);
    int b = (int)(idx / ((long long)IMG_W * IMG_H));

    const float FOC = 320.0f / tanf(0.017453292519943295f * 39.5f);
    float depth = obs[(((long long)b * NCH + 3) * IMG_H + h) * IMG_W + w];
    float gx = (float)w;
    float gz = (float)(IMG_H - 1 - h);
    float Xp = (gx - XCAMF) * depth / FOC;
    float Zp = (gz - ZCAMF) * depth / FOC;
    float a  = view_angles[b] * 0.017453292519943295f;
    float ca = __cosf(a), sa = __sinf(a);
    float Yv = ca * depth - sa * Zp;
    float Zv = sa * depth + ca * Zp + AGENT_HF;
    float Xv = Xp + 250.0f;                       // + VR*RES/2
    float xs = (Xv / 5.0f - 50.0f) / 100.0f * 2.0f;
    float ys = (Yv / 5.0f - 50.0f) / 100.0f * 2.0f;
    float zs = (Zv / 5.0f - 32.0f) / 80.0f * 2.0f;

    float pos[3] = { xs * 50.0f + 50.0f, ys * 50.0f + 50.0f, zs * 40.0f + 40.0f };
    const int gd[3] = { VRD, VRD, ZHD };
    float wts[3][2];
    int   ip[3][2];
#pragma unroll
    for (int d = 0; d < 3; ++d) {
        float fl = floorf(pos[d]);
#pragma unroll
        for (int k = 0; k < 2; ++k) {
            float pix = fl + (float)k;
            bool safe = (pix > 0.0f) && (pix < (float)gd[d]);
            wts[d][k] = safe ? (1.0f - fabsf(pos[d] - pix)) : 0.0f;
            ip[d][k]  = safe ? (int)pix : 0;
        }
    }

    float feat[17];
    feat[0] = 1.0f;
#pragma unroll
    for (int c = 0; c < NCAT; ++c)
        feat[c + 1] = obs[(((long long)b * NCH + 4 + c) * IMG_H + h) * IMG_W + w];

    const long long cstride = (long long)VRD * VRD * ZHD;
#pragma unroll
    for (int k0 = 0; k0 < 2; ++k0)
#pragma unroll
        for (int k1 = 0; k1 < 2; ++k1)
#pragma unroll
            for (int k2 = 0; k2 < 2; ++k2) {
                float wgt = wts[0][k0] * wts[1][k1] * wts[2][k2];
                if (wgt == 0.0f) continue;
                int xg = ip[0][k0], yg = ip[1][k1], zg = ip[2][k2];
                long long base = ((((long long)b * 17) * VRD + yg) * VRD + xg) * ZHD + zg;
                for (int c = 0; c < 17; ++c)
                    atomicAdd(&vox[base + (long long)c * cstride], feat[c] * wgt);
            }
}

// =====================================================================
// Kernel 3: WMMA z-reduction.  One wave handles 16 (y,x) cells of one (b,c).
// B[k][n] = rounded voxel z-value k for cell n; A = ones -> D[m][n] = sum_k B[k][n].
// B k order used here (contiguous): lane n (0..15) holds K=0..15 of column n,
// lane n+16 holds K=16..31 of column n (k ordered low->high across the v16h).
// Two accumulators: windowed z in [9,49), full z in [0,80).
// z-values per (lane,chunk) are a contiguous, 16B-aligned run of 16 floats
// (cells are 80 floats = 320 B apart), so load them with four float4
// (global_load_b128) and hoist the only possible OOB case (chunk 2, upper
// lane half: z0 = 80 >= 80) to a single group-level branch.
// =====================================================================
__global__ void sm_zreduce_wmma_kernel(const float* __restrict__ vox,
                                       float* __restrict__ preds,
                                       float* __restrict__ out_fp) {
    const int TILES = BSZ * 17 * 625;                 // 625 = 10000/16 cells
    int wave = (int)(((long long)blockIdx.x * blockDim.x + threadIdx.x) >> 5);
    int lane = threadIdx.x & 31;
    if (wave >= TILES) return;                        // wave-uniform exit

    int bc = wave / 625;
    int t  = wave % 625;
    int b  = bc / 17;
    int c  = bc % 17;
    int cell = t * 16 + (lane & 15);
    int y = cell / VRD, x = cell % VRD;
    const float* vp = vox + ((((long long)b * 17 + c) * VRD + y) * VRD + x) * (long long)ZHD;
    int zoff = (lane >= 16) ? 16 : 0;

    v16h ones;
#pragma unroll
    for (int i = 0; i < 16; ++i) ones[i] = (_Float16)1.0f;

    v8f accW = {};   // windowed sum
    v8f accF = {};   // full sum
#pragma unroll
    for (int chunk = 0; chunk < 3; ++chunk) {
        int z0 = chunk * 32 + zoff;
        float vals[16];
        if (z0 < ZHD) {                               // whole 16-run in range
            const float4* p = (const float4*)(vp + z0);
            float4 q0 = p[0], q1 = p[1], q2 = p[2], q3 = p[3];
            vals[ 0] = q0.x; vals[ 1] = q0.y; vals[ 2] = q0.z; vals[ 3] = q0.w;
            vals[ 4] = q1.x; vals[ 5] = q1.y; vals[ 6] = q1.z; vals[ 7] = q1.w;
            vals[ 8] = q2.x; vals[ 9] = q2.y; vals[10] = q2.z; vals[11] = q2.w;
            vals[12] = q3.x; vals[13] = q3.y; vals[14] = q3.z; vals[15] = q3.w;
        } else {                                      // chunk 2, upper half: zero pad
#pragma unroll
            for (int i = 0; i < 16; ++i) vals[i] = 0.0f;
        }
        v16h bF, bW;
#pragma unroll
        for (int i = 0; i < 16; ++i) {
            int z = z0 + i;
            float v = rintf(vals[i]);                 // jnp.round (RNE)
            bF[i] = (_Float16)v;
            bW[i] = (_Float16)((z >= MINZI && z < MAXZI) ? v : 0.0f);
        }
        accW = __builtin_amdgcn_wmma_f32_16x16x32_f16(false, ones, false, bW,
                                                      (short)0, accW, false, false);
        accF = __builtin_amdgcn_wmma_f32_16x16x32_f16(false, ones, false, bF,
                                                      (short)0, accF, false, false);
    }
    // C/D layout: VGPR0 holds row M=0 (lanes 0-15, N=lane) and M=8 (lanes 16-31).
    // All rows equal since A is all-ones, so acc[0] is the column sum on every lane.
    float sumW = accW[0];
    float sumF = accF[0];

    if (lane < 16) {
        if (c == 0) {
            float fp = fminf(fmaxf(sumW, 0.0f), 1.0f);          // MAP_THR = 1
            preds[(((long long)b * 18 + 0) * VRD + y) * VRD + x] = fp;
            out_fp[((long long)b * VRD + y) * VRD + x] = fp;    // output 0
        } else {
            float cat = fminf(fmaxf(sumW * 0.2f, 0.0f), 1.0f);  // CAT_THR = 5
            preds[(((long long)b * 18 + (c + 1)) * VRD + y) * VRD + x] = cat;
        }
    } else if (c == 0) {
        float ex = fminf(fmaxf(sumF, 0.0f), 1.0f);              // EXP_THR = 1
        preds[(((long long)b * 18 + 1) * VRD + y) * VRD + x] = ex;
    }
}

// =====================================================================
// Kernel 4: rotation grid_sample (agent_view is virtual: crop of preds, rest 0)
// =====================================================================
__device__ __forceinline__ float sm_agent_view(const float* __restrict__ preds,
                                               int b, int c, int yi, int xi) {
    if (yi < CY1 || yi >= CY2 || xi < CX1 || xi >= CX2) return 0.0f;
    int pc;
    if (c == 0) pc = 0;
    else if (c == 1) pc = 1;
    else if (c >= 4) pc = c - 2;   // cat channels 4..19 -> preds 2..17
    else return 0.0f;              // channels 2,3 never written
    return preds[(((long long)b * 18 + pc) * VRD + (yi - CY1)) * VRD + (xi - CX1)];
}

__global__ void sm_rotate_kernel(const float* __restrict__ params,
                                 const float* __restrict__ preds,
                                 float* __restrict__ rotated) {
    long long idx = (long long)blockIdx.x * blockDim.x + threadIdx.x;
    const long long total = (long long)BSZ * NCH * MDIM * MDIM;
    if (idx >= total) return;
    int j = (int)(idx % MDIM);
    int i = (int)((idx / MDIM) % MDIM);
    int c = (int)((idx / ((long long)MDIM * MDIM)) % NCH);
    int b = (int)(idx / ((long long)MDIM * MDIM * NCH));

    float ct = params[b * 4 + 0], st = params[b * 4 + 1];
    float Xg = -1.0f + 2.0f * (float)j / (float)(MDIM - 1);
    float Yg = -1.0f + 2.0f * (float)i / (float)(MDIM - 1);
    float gxv = ct * Xg - st * Yg;
    float gyv = st * Xg + ct * Yg;
    float xf = (gxv + 1.0f) * 0.5f * (float)(MDIM - 1);
    float yf = (gyv + 1.0f) * 0.5f * (float)(MDIM - 1);
    float x0 = floorf(xf), y0 = floorf(yf);
    float wx1 = xf - x0, wy1 = yf - y0;

    float acc = 0.0f;
#pragma unroll
    for (int dy = 0; dy < 2; ++dy)
#pragma unroll
        for (int dx = 0; dx < 2; ++dx) {
            float ix = x0 + (float)dx, iy = y0 + (float)dy;
            if (ix < 0.0f || ix > (float)(MDIM - 1) ||
                iy < 0.0f || iy > (float)(MDIM - 1)) continue;
            float wq = (dx ? wx1 : 1.0f - wx1) * (dy ? wy1 : 1.0f - wy1);
            acc += wq * sm_agent_view(preds, b, c, (int)iy, (int)ix);
        }
    rotated[idx] = acc;
}

// =====================================================================
// Kernel 5: translation grid_sample + fused max with maps_last
// =====================================================================
__global__ void sm_translate_kernel(const float* __restrict__ params,
                                    const float* __restrict__ rotated,
                                    const float* __restrict__ maps_last,
                                    float* __restrict__ out) {
    long long idx = (long long)blockIdx.x * blockDim.x + threadIdx.x;
    const long long total = (long long)BSZ * NCH * MDIM * MDIM;
    if (idx >= total) return;
    int j = (int)(idx % MDIM);
    int i = (int)((idx / MDIM) % MDIM);
    int c = (int)((idx / ((long long)MDIM * MDIM)) % NCH);
    int b = (int)(idx / ((long long)MDIM * MDIM * NCH));

    float stx = params[b * 4 + 2], sty = params[b * 4 + 3];
    float Xg = -1.0f + 2.0f * (float)j / (float)(MDIM - 1);
    float Yg = -1.0f + 2.0f * (float)i / (float)(MDIM - 1);
    float gxv = Xg + stx;
    float gyv = Yg + sty;
    float xf = (gxv + 1.0f) * 0.5f * (float)(MDIM - 1);
    float yf = (gyv + 1.0f) * 0.5f * (float)(MDIM - 1);
    float x0 = floorf(xf), y0 = floorf(yf);
    float wx1 = xf - x0, wy1 = yf - y0;

    const long long plane = (long long)MDIM * MDIM;
    const float* src = rotated + ((long long)b * NCH + c) * plane;
    float acc = 0.0f;
#pragma unroll
    for (int dy = 0; dy < 2; ++dy)
#pragma unroll
        for (int dx = 0; dx < 2; ++dx) {
            float ix = x0 + (float)dx, iy = y0 + (float)dy;
            if (ix < 0.0f || ix > (float)(MDIM - 1) ||
                iy < 0.0f || iy > (float)(MDIM - 1)) continue;
            float wq = (dx ? wx1 : 1.0f - wx1) * (dy ? wy1 : 1.0f - wy1);
            acc += wq * src[(long long)(int)iy * MDIM + (int)ix];
        }
    out[O_TRANS + idx] = acc;                           // output 4: translated
    out[O_MAP + idx]   = fmaxf(maps_last[idx], acc);    // output 1: map_pred
}

// =====================================================================
extern "C" void kernel_launch(void* const* d_in, const int* in_sizes, int n_in,
                              void* d_out, int out_size, void* d_ws, size_t ws_size,
                              hipStream_t stream) {
    (void)in_sizes; (void)n_in; (void)out_size;
    const float* obs         = (const float*)d_in[0];
    const float* pose_obs    = (const float*)d_in[1];
    const float* maps_last   = (const float*)d_in[2];
    const float* poses_last  = (const float*)d_in[3];
    const float* view_angles = (const float*)d_in[4];
    float* out = (float*)d_out;
    float* ws  = (float*)d_ws;

    const size_t need_bytes = (size_t)(ROT_OFF + ROT_SIZE) * sizeof(float);
    if (ws_size < need_bytes) return;  // not enough scratch: bail deterministically

    // zero voxel grid + preds (+ params) — graph-capture-safe stream op
    hipMemsetAsync(d_ws, 0, (size_t)(VOX_OFF + VOX_SIZE) * sizeof(float), stream);

    sm_pose_kernel<<<1, 32, 0, stream>>>(pose_obs, poses_last, out, ws + PARAMS_OFF);

    const long long npix = (long long)BSZ * IMG_H * IMG_W;
    sm_splat_kernel<<<(int)((npix + 255) / 256), 256, 0, stream>>>(
        obs, view_angles, ws + VOX_OFF);

    const int tiles = BSZ * 17 * 625;
    const long long rthreads = (long long)tiles * 32;
    sm_zreduce_wmma_kernel<<<(int)((rthreads + 255) / 256), 256, 0, stream>>>(
        ws + VOX_OFF, ws + PREDS_OFF, out + O_FP);

    const long long nmap = (long long)BSZ * NCH * MDIM * MDIM;
    sm_rotate_kernel<<<(int)((nmap + 255) / 256), 256, 0, stream>>>(
        ws + PARAMS_OFF, ws + PREDS_OFF, ws + ROT_OFF);

    sm_translate_kernel<<<(int)((nmap + 255) / 256), 256, 0, stream>>>(
        ws + PARAMS_OFF, ws + ROT_OFF, maps_last, out);
}
